// RNNEnc_81355270520885
// MI455X (gfx1250) — compile-verified, matrix operand order checked by reading, not run on hardware
//
#include <hip/hip_runtime.h>
#include <hip/hip_bf16.h>
#include <stdint.h>

// CDNA5 / gfx1250 bf16 WMMA bidirectional GRU encoder.
// [convert fp32->bf16] -> [WMMA GEMM gi = x@W_ih^T + b_ih, weights TDM-staged in LDS]
// -> [4-WGP recurrence: W_hh register-resident, gh via WMMA, fused gates]
// -> [assemble output slice].

typedef __attribute__((ext_vector_type(16))) __bf16 v16bf;
typedef __attribute__((ext_vector_type(8)))  __bf16 v8bf;
typedef __attribute__((ext_vector_type(8)))  float  v8f;
typedef __attribute__((ext_vector_type(4)))  unsigned int u32x4;
typedef __attribute__((ext_vector_type(8)))  int i32x8;
typedef __attribute__((ext_vector_type(4)))  int i32x4;

#define B_  32
#define S_  2048
#define D_  256
#define G_  768   // 3*D

#if defined(__has_builtin)
#if __has_builtin(__builtin_amdgcn_tensor_load_to_lds) && \
    __has_builtin(__builtin_amdgcn_s_wait_tensorcnt)
#define USE_TDM 1
#endif
#endif

union ABu { v16bf v; v8bf h[2]; };

__device__ __forceinline__ float fast_sigmoid(float x) {
  return 1.0f / (1.0f + __expf(-x));
}
__device__ __forceinline__ float fast_tanh(float x) {
  float e = __expf(-2.0f * x);
  return (1.0f - e) / (1.0f + e);
}

// ---------------- conversion kernels ----------------
__global__ void k_convert_x(const float* __restrict__ x, __bf16* __restrict__ xb, int n) {
  int i = blockIdx.x * blockDim.x + threadIdx.x;
  if (i < n) xb[i] = (__bf16)x[i];
}

__global__ void k_convert_w(const float* __restrict__ w0, const float* __restrict__ w1,
                            const float* __restrict__ w2, const float* __restrict__ w3,
                            __bf16* __restrict__ o0, __bf16* __restrict__ o1,
                            __bf16* __restrict__ o2, __bf16* __restrict__ o3) {
  int i = blockIdx.x * blockDim.x + threadIdx.x;
  if (i < G_ * D_) {
    o0[i] = (__bf16)w0[i];
    o1[i] = (__bf16)w1[i];
    o2[i] = (__bf16)w2[i];
    o3[i] = (__bf16)w3[i];
  }
}

// ---------------- kernel A: gi = x @ W_ih^T + b_ih (both dirs) ----------------
// Block = 8 waves sharing one 64-column N-group; the 64x256 bf16 weight slab
// (32KB) is staged into LDS via the Tensor Data Mover. Each wave: 16x64 output
// tile, A half-row preloaded into 64 VGPRs, B streamed from LDS.
__global__ __launch_bounds__(256)
void k_gemm_gi(const __bf16* __restrict__ xb,
               const __bf16* __restrict__ Wf, const __bf16* __restrict__ Wb,
               const float* __restrict__ bias_f, const float* __restrict__ bias_b,
               float* __restrict__ gi_f, float* __restrict__ gi_b) {
  const int dir = blockIdx.z;
  const __bf16* W    = dir ? Wb : Wf;
  const float*  bias = dir ? bias_b : bias_f;
  float*        gi   = dir ? gi_b : gi_f;

  __shared__ __bf16 lw[64 * D_];            // 32KB weight slab for this n-group

  const int ng = blockIdx.y;                // n-base = ng*64
  const __bf16* wsrc = W + (size_t)ng * 64 * D_;

#ifdef USE_TDM
  if (threadIdx.x == 0) {
    const unsigned long long ga = (unsigned long long)(uintptr_t)wsrc;
    const unsigned int ldsoff   = (unsigned int)(uintptr_t)(&lw[0]); // addr[31:0] = LDS offset
    u32x4 g0;
    g0[0] = 1u;                                             // count=1 (valid), user mode
    g0[1] = ldsoff;                                         // lds_addr
    g0[2] = (unsigned int)(ga & 0xFFFFFFFFu);               // global_addr[31:0]
    g0[3] = (unsigned int)((ga >> 32) & 0x01FFFFFFu)        // global_addr[56:32]
          | (2u << 30);                                     // type = 2 ("image")
    i32x8 g1;
    g1[0] = (int)(1u << 16);      // workgroup_mask=0, data_size=1 (2 bytes)
    g1[1] = (int)(256u << 16);    // tensor_dim0 = 256 (K elements per row)
    g1[2] = (int)(64u << 16);     // tensor_dim1 = 64 rows
    g1[3] = (int)(256u << 16);    // tile_dim0 = 256
    g1[4] = 64;                   // tile_dim1 = 64, tile_dim2 = 0
    g1[5] = 256;                  // tensor_dim0_stride = 256
    g1[6] = 0;
    g1[7] = 0;
    i32x4 z4 = {0, 0, 0, 0};
    i32x8 z8 = {0, 0, 0, 0, 0, 0, 0, 0};
    __builtin_amdgcn_tensor_load_to_lds(g0, g1, z4, z4, z8, 0);  // 6-arg form
  }
  __builtin_amdgcn_s_wait_tensorcnt(0);     // only issuing wave has TENSORcnt>0
#else
  for (int c = threadIdx.x; c < (64 * D_) / 8; c += 256)
    *(v8bf*)(lw + c * 8) = *(const v8bf*)(wsrc + c * 8);
#endif
  __syncthreads();

  const int wave = threadIdx.x >> 5;
  const int lane = threadIdx.x & 31;
  const int half = (lane >> 4) & 1;
  const int r    = lane & 15;

  const int mt   = blockIdx.x * 8 + wave;   // 0..4095 : 16-row tile of flattened (b,t)
  const int row0 = mt * 16;
  const int b    = row0 >> 11;              // / S_
  const int t0   = row0 & (S_ - 1);
  const int t    = t0 + r;                  // this lane's A row (M = r)
  const int xrow = b * S_ + (dir ? (S_ - 1 - t) : t);
  const __bf16* xr = xb + (size_t)xrow * D_;

  // preload this lane's full A half-row (8 K-fragments, 64 VGPRs)
  ABu afrag[8];
#pragma unroll
  for (int kk = 0; kk < 8; ++kk) {
    const int k0 = kk * 32;
    afrag[kk].h[0] = *(const v8bf*)(xr + k0 + half * 8);
    afrag[kk].h[1] = *(const v8bf*)(xr + k0 + 16 + half * 8);
  }

  v8f acc[4] = {};
#pragma unroll
  for (int kk = 0; kk < 8; ++kk) {
    const int k0 = kk * 32;
#pragma unroll
    for (int j = 0; j < 4; ++j) {
      const int nl = j * 16 + r;            // local row in weight slab
      ABu bm;
      bm.h[0] = *(const v8bf*)(&lw[nl * D_ + k0 + half * 8]);
      bm.h[1] = *(const v8bf*)(&lw[nl * D_ + k0 + 16 + half * 8]);
      acc[j] = __builtin_amdgcn_wmma_f32_16x16x32_bf16(
          false, afrag[kk].v, false, bm.v, (short)0, acc[j], false, false);
    }
  }

#pragma unroll
  for (int j = 0; j < 4; ++j) {
    const int n  = ng * 64 + j * 16 + r;    // D layout: lane -> column, vgpr i -> row i+half*8
    const float bv = bias[n];
#pragma unroll
    for (int i = 0; i < 8; ++i) {
      const int m = i + half * 8;
      gi[(size_t)(row0 + m) * G_ + n] = acc[j][i] + bv;
    }
  }
}

// ---------------- kernel B: sequential GRU recurrence ----------------
// grid = (2 batch-halves, 2 directions); block = 512 threads = 16 waves.
// W_hh fragments (3 N-tiles x 8 K-steps per wave = 192 VGPRs) are hoisted out
// of the time loop: per step only LDS h-loads + 24 WMMAs + fused gates.
__global__ __launch_bounds__(512)
void k_recurrence(const __bf16* __restrict__ Whh_f, const __bf16* __restrict__ Whh_b,
                  const float* __restrict__ gi_f,  const float* __restrict__ gi_b,
                  const float* __restrict__ bhh_f, const float* __restrict__ bhh_b,
                  float* __restrict__ hf_hist, float* __restrict__ hb_hist) {
  const int dir = blockIdx.y;
  const int b0  = blockIdx.x * 16;
  const __bf16* W  = dir ? Whh_b : Whh_f;
  const float*  gi = dir ? gi_b  : gi_f;
  const float*  bh = dir ? bhh_b : bhh_f;
  float*  hist     = dir ? hb_hist : hf_hist;

  __shared__ __bf16 lh_bf[16 * D_];   // 8 KB  : h as WMMA A operand (row-major bf16)
  __shared__ float  lh_f [16 * D_];   // 16 KB : h fp32 master copy
  __shared__ float  lgh  [G_ * 16];   // 48 KB : gh, stored [n][m]

  const int tid = threadIdx.x;
  for (int i = tid; i < 16 * D_; i += 512) { lh_bf[i] = (__bf16)0.0f; lh_f[i] = 0.0f; }

  const int wave = tid >> 5;
  const int lane = tid & 31;
  const int half = (lane >> 4) & 1;
  const int r    = lane & 15;

  // hoist W_hh fragments (time-invariant) into registers
  ABu wfrag[3][8];
#pragma unroll
  for (int jt = 0; jt < 3; ++jt) {
    const int n0 = (wave * 3 + jt) * 16;
    const __bf16* wr = W + (size_t)(n0 + r) * D_;
#pragma unroll
    for (int kk = 0; kk < 8; ++kk) {
      const int k0 = kk * 32;
      wfrag[jt][kk].h[0] = *(const v8bf*)(wr + k0 + half * 8);
      wfrag[jt][kk].h[1] = *(const v8bf*)(wr + k0 + 16 + half * 8);
    }
  }
  __syncthreads();

  // gate-phase mapping: 8 consecutive units per thread (coalesced gi reads)
  const int qm = (tid * 8) >> 8;      // batch row within tile
  const int qj = (tid * 8) & 255;     // first hidden unit

  for (int t = 0; t < S_; ++t) {
    // --- gh = h @ W_hh^T : A from LDS, B from registers ---
    v8f acc[3] = {};
#pragma unroll
    for (int kk = 0; kk < 8; ++kk) {
      const int k0 = kk * 32;
      ABu a;
      a.h[0] = *(const v8bf*)(&lh_bf[r * D_ + k0 + half * 8]);
      a.h[1] = *(const v8bf*)(&lh_bf[r * D_ + k0 + 16 + half * 8]);
#pragma unroll
      for (int jt = 0; jt < 3; ++jt)
        acc[jt] = __builtin_amdgcn_wmma_f32_16x16x32_bf16(
            false, a.v, false, wfrag[jt][kk].v, (short)0, acc[jt], false, false);
    }
#pragma unroll
    for (int jt = 0; jt < 3; ++jt) {
      const int n = (wave * 3 + jt) * 16 + r;
#pragma unroll
      for (int i = 0; i < 8; ++i) lgh[n * 16 + i + half * 8] = acc[jt][i];
    }
    __syncthreads();

    // --- fused gate math; writes next h into LDS + history to global ---
    {
      const size_t girow = ((size_t)(b0 + qm) * S_ + t) * G_;
#pragma unroll
      for (int e = 0; e < 8; ++e) {
        const int j = qj + e;
        const float i_r = gi[girow + j];
        const float i_z = gi[girow + 256 + j];
        const float i_n = gi[girow + 512 + j];
        const float h_r = lgh[j * 16 + qm]         + bh[j];
        const float h_z = lgh[(256 + j) * 16 + qm] + bh[256 + j];
        const float h_n = lgh[(512 + j) * 16 + qm] + bh[512 + j];
        const float rg = fast_sigmoid(i_r + h_r);
        const float zg = fast_sigmoid(i_z + h_z);
        const float nn = fast_tanh(i_n + rg * h_n);
        const float hp = lh_f[qm * D_ + j];
        const float hv = (1.0f - zg) * nn + zg * hp;
        lh_f [qm * D_ + j] = hv;
        lh_bf[qm * D_ + j] = (__bf16)hv;
        hist[((size_t)(b0 + qm) * S_ + t) * D_ + j] = hv;
        __builtin_prefetch(&gi[girow + G_ + j], 0, 1);  // next step's gi row
      }
    }
    __syncthreads();
  }
}

// ---------------- kernel C: output assembly ----------------
__global__ void k_assemble(const float* __restrict__ v_in,
                           const float* __restrict__ hf, const float* __restrict__ hb,
                           float* __restrict__ out, int cl, int n_out) {
  int i = blockIdx.x * blockDim.x + threadIdx.x;
  if (i >= n_out) return;
  const int Sout = S_ - 2 * cl;
  const int c   = i & 511;
  const int rem = i >> 9;
  const int tp  = rem % Sout;
  const int b   = rem / Sout;
  const int t   = tp + cl;
  if (c < D_) {
    out[i] = hf[((size_t)b * S_ + t) * D_ + c] + v_in[((size_t)b * S_ + t) * D_ + c];
  } else {
    const int d = c - D_;
    out[i] = hb[((size_t)b * S_ + t) * D_ + d] +
             v_in[((size_t)b * S_ + (S_ - 1 - t)) * D_ + d];
  }
}

// ---------------- host launcher ----------------
extern "C" void kernel_launch(void* const* d_in, const int* in_sizes, int n_in,
                              void* d_out, int out_size, void* d_ws, size_t ws_size,
                              hipStream_t stream) {
  (void)in_sizes; (void)n_in; (void)ws_size;

  const float* v_in   = (const float*)d_in[0];
  const float* w_ih_f = (const float*)d_in[1];
  const float* w_hh_f = (const float*)d_in[2];
  const float* b_ih_f = (const float*)d_in[3];
  const float* b_hh_f = (const float*)d_in[4];
  const float* w_ih_b = (const float*)d_in[5];
  const float* w_hh_b = (const float*)d_in[6];
  const float* b_ih_b = (const float*)d_in[7];
  const float* b_hh_b = (const float*)d_in[8];
  // d_in[9] = context_length (device scalar); derive deterministically from out_size
  const int Sout = out_size / (B_ * 2 * D_);
  const int cl   = (S_ - Sout) / 2;

  // workspace carve-up
  char* ws = (char*)d_ws;
  float* gi_f    = (float*)ws; ws += (size_t)B_ * S_ * G_ * 4;
  float* gi_b    = (float*)ws; ws += (size_t)B_ * S_ * G_ * 4;
  float* hf_hist = (float*)ws; ws += (size_t)B_ * S_ * D_ * 4;
  float* hb_hist = (float*)ws; ws += (size_t)B_ * S_ * D_ * 4;
  __bf16* x_bf   = (__bf16*)ws; ws += (size_t)B_ * S_ * D_ * 2;
  __bf16* wihf   = (__bf16*)ws; ws += (size_t)G_ * D_ * 2;
  __bf16* whhf   = (__bf16*)ws; ws += (size_t)G_ * D_ * 2;
  __bf16* wihb   = (__bf16*)ws; ws += (size_t)G_ * D_ * 2;
  __bf16* whhb   = (__bf16*)ws; ws += (size_t)G_ * D_ * 2;

  { const int n = B_ * S_ * D_;
    k_convert_x<<<(n + 255) / 256, 256, 0, stream>>>(v_in, x_bf, n); }
  k_convert_w<<<(G_ * D_ + 255) / 256, 256, 0, stream>>>(
      w_ih_f, w_hh_f, w_ih_b, w_hh_b, wihf, whhf, wihb, whhb);

  dim3 gA(512, 12, 2);  // 4096 M-tiles / 8 waves, 12 N-groups of 64, 2 dirs
  k_gemm_gi<<<gA, 256, 0, stream>>>(x_bf, wihf, wihb, b_ih_f, b_ih_b, gi_f, gi_b);

  dim3 gB(2, 2);        // 2 batch-halves x 2 directions
  k_recurrence<<<gB, 512, 0, stream>>>(whhf, whhb, gi_f, gi_b,
                                       b_hh_f, b_hh_b, hf_hist, hb_hist);

  k_assemble<<<(out_size + 255) / 256, 256, 0, stream>>>(
      v_in, hf_hist, hb_hist, (float*)d_out, cl, out_size);
}
